// MixtralAttention_42949672960146
// MI455X (gfx1250) — compile-verified, matrix operand order checked by reading, not run on hardware
//
#include <hip/hip_runtime.h>
#include <hip/hip_bf16.h>

typedef __attribute__((ext_vector_type(16))) _Float16 v16h;
typedef __attribute__((ext_vector_type(8)))  _Float16 v8h;
typedef __attribute__((ext_vector_type(4)))  _Float16 v4h;
typedef __attribute__((ext_vector_type(2)))  _Float16 v2h;
typedef __attribute__((ext_vector_type(8)))  float    v8f;

#define S_LEN 2048
#define H_DIM 4096
#define NHEAD 32
#define NKVH  8
#define HDIM  128

__device__ inline v8f wmma_f16(v16h a, v16h b, v8f c) {
  return __builtin_amdgcn_wmma_f32_16x16x32_f16(
      false, a, false, b, (short)0, c, false, false);
}

__device__ inline v16h cat8(v8h a, v8h b) {
  return __builtin_shufflevector(a, b, 0, 1, 2, 3, 4, 5, 6, 7,
                                 8, 9, 10, 11, 12, 13, 14, 15);
}
// memory holds [lo0,hi0,lo1,hi1,...]; fragment wants [lo0..lo7,hi0..hi7]
__device__ inline v16h deinterleave(v8h a, v8h b) {
  return __builtin_shufflevector(a, b, 0, 2, 4, 6, 8, 10, 12, 14,
                                 1, 3, 5, 7, 9, 11, 13, 15);
}

// ---------------- RMSNorm -> f16 ----------------
__global__ __launch_bounds__(256) void rmsnorm_kernel(
    const float* __restrict__ x, const float* __restrict__ w,
    _Float16* __restrict__ h) {
  int row = blockIdx.x;
  const float* xr = x + (size_t)row * H_DIM;
  float ss = 0.f;
  for (int c = threadIdx.x * 4; c < H_DIM; c += 1024) {
    float4 v = *(const float4*)(xr + c);
    ss += v.x * v.x + v.y * v.y + v.z * v.z + v.w * v.w;
  }
  __shared__ float red[256];
  red[threadIdx.x] = ss;
  __syncthreads();
  for (int s2 = 128; s2 > 0; s2 >>= 1) {
    if (threadIdx.x < s2) red[threadIdx.x] += red[threadIdx.x + s2];
    __syncthreads();
  }
  float scale = rsqrtf(red[0] * (1.f / H_DIM) + 1e-5f);
  for (int c = threadIdx.x * 4; c < H_DIM; c += 1024) {
    float4 v = *(const float4*)(xr + c);
    float4 g = *(const float4*)(w + c);
    v4h o;
    o[0] = (_Float16)(v.x * scale * g.x);
    o[1] = (_Float16)(v.y * scale * g.y);
    o[2] = (_Float16)(v.z * scale * g.z);
    o[3] = (_Float16)(v.w * scale * g.w);
    *(v4h*)(h + (size_t)row * H_DIM + c) = o;
  }
}

// ------- Tiled WMMA GEMM: C[M,N] = A_f16[M,K] * B_f32[K,N] (+resid), 128x128x32 -------
#define TM 128
#define TN 128
#define TK 32
#define LDSTR 40  // padded stride in halfs (16B-aligned rows/cols)

__global__ __launch_bounds__(256) void gemm_wmma(
    const _Float16* __restrict__ A, const float* __restrict__ Bw,
    float* __restrict__ C, const float* __restrict__ resid,
    int M, int N, int K) {
  __shared__ alignas(16) _Float16 As[TM * LDSTR];  // row-major  [row][k]
  __shared__ alignas(16) _Float16 Bs[TN * LDSTR];  // col-major  [col][k]
  int bm = blockIdx.y * TM;
  int bn = blockIdx.x * TN;
  int t = threadIdx.x;
  int wave = t >> 5, lane = t & 31;
  int wr = wave & 3;   // rows wr*32 .. +31  (2 fragment rows)
  int wc = wave >> 2;  // cols wc*64 .. +63  (4 fragment cols)
  int half = lane >> 4, l16 = lane & 15;

  v8f acc[2][4];
#pragma unroll
  for (int i = 0; i < 2; ++i)
#pragma unroll
    for (int j = 0; j < 4; ++j) acc[i][j] = (v8f){};

  for (int k0 = 0; k0 < K; k0 += TK) {
    { // A tile 128x32 f16: two 16B global loads + two 16B LDS stores
      int row = t >> 1;
      int cs = (t & 1) * 16;
      const _Float16* src = A + (size_t)(bm + row) * K + k0 + cs;
      *(v8h*)&As[row * LDSTR + cs]     = *(const v8h*)src;
      *(v8h*)&As[row * LDSTR + cs + 8] = *(const v8h*)(src + 8);
    }
    { // B tile 32x128 f32 -> f16, col-major: 4x4 patch, 4 b128 loads + 4 b64 stores
      int c4 = (t & 31) * 4;   // 4 columns
      int kk = (t >> 5) * 4;   // 4 k values
      const float* srcB = Bw + (size_t)(k0 + kk) * N + bn + c4;
      float4 r0 = *(const float4*)srcB;
      float4 r1 = *(const float4*)(srcB + N);
      float4 r2 = *(const float4*)(srcB + 2 * (size_t)N);
      float4 r3 = *(const float4*)(srcB + 3 * (size_t)N);
      v4h c0, c1, c2, c3;
      c0[0] = (_Float16)r0.x; c0[1] = (_Float16)r1.x; c0[2] = (_Float16)r2.x; c0[3] = (_Float16)r3.x;
      c1[0] = (_Float16)r0.y; c1[1] = (_Float16)r1.y; c1[2] = (_Float16)r2.y; c1[3] = (_Float16)r3.y;
      c2[0] = (_Float16)r0.z; c2[1] = (_Float16)r1.z; c2[2] = (_Float16)r2.z; c2[3] = (_Float16)r3.z;
      c3[0] = (_Float16)r0.w; c3[1] = (_Float16)r1.w; c3[2] = (_Float16)r2.w; c3[3] = (_Float16)r3.w;
      *(v4h*)&Bs[(c4 + 0) * LDSTR + kk] = c0;
      *(v4h*)&Bs[(c4 + 1) * LDSTR + kk] = c1;
      *(v4h*)&Bs[(c4 + 2) * LDSTR + kk] = c2;
      *(v4h*)&Bs[(c4 + 3) * LDSTR + kk] = c3;
    }
    __syncthreads();

    v16h af[2], bf[4];
#pragma unroll
    for (int i = 0; i < 2; ++i) {
      const _Float16* arow = &As[(wr * 32 + i * 16 + l16) * LDSTR];
      af[i] = cat8(*(const v8h*)(arow + half * 8),
                   *(const v8h*)(arow + 16 + half * 8));
    }
#pragma unroll
    for (int j = 0; j < 4; ++j) {
      const _Float16* bcol = &Bs[(wc * 64 + j * 16 + l16) * LDSTR];
      bf[j] = cat8(*(const v8h*)(bcol + half * 8),
                   *(const v8h*)(bcol + 16 + half * 8));
    }
#pragma unroll
    for (int i = 0; i < 2; ++i)
#pragma unroll
      for (int j = 0; j < 4; ++j)
        acc[i][j] = wmma_f16(af[i], bf[j], acc[i][j]);
    __syncthreads();
  }

  // ---- epilogue: residual branch hoisted once (uniform), pointer-walked rows ----
  {
    int rowg0 = bm + wr * 32 + half * 8;
    int colg0 = bn + wc * 64 + l16;
    if (resid != nullptr) {
#pragma unroll
      for (int i = 0; i < 2; ++i) {
        size_t base = (size_t)(rowg0 + i * 16) * N + colg0;
#pragma unroll
        for (int r = 0; r < 8; ++r) {
          float* crow = C + base + (size_t)r * N;
          const float* rrow = resid + base + (size_t)r * N;
#pragma unroll
          for (int j = 0; j < 4; ++j)
            crow[j * 16] = acc[i][j][r] + rrow[j * 16];
        }
      }
    } else {
#pragma unroll
      for (int i = 0; i < 2; ++i) {
        size_t base = (size_t)(rowg0 + i * 16) * N + colg0;
#pragma unroll
        for (int r = 0; r < 8; ++r) {
          float* crow = C + base + (size_t)r * N;
#pragma unroll
          for (int j = 0; j < 4; ++j)
            crow[j * 16] = acc[i][j][r];
        }
      }
    }
  }
}

// ---------------- RoPE (fp32 in, f16 out) ----------------
__global__ __launch_bounds__(256) void rope_kernel(
    const float* __restrict__ qin, const float* __restrict__ cosb,
    const float* __restrict__ sinb, _Float16* __restrict__ qout, int nheads) {
  int idx = blockIdx.x * 256 + threadIdx.x;  // S*nheads*64
  int d = idx & 63;
  int h = (idx >> 6) % nheads;
  int s = idx / (64 * nheads);
  float c = cosb[s * HDIM + d];
  float sn = sinb[s * HDIM + d];
  size_t base = ((size_t)s * nheads + h) * HDIM;
  float x1 = qin[base + d], x2 = qin[base + d + 64];
  qout[base + d]      = (_Float16)(x1 * c - x2 * sn);
  qout[base + d + 64] = (_Float16)(x2 * c + x1 * sn);
}

// V [S, NKV, HD] f32 -> V^T [NKV, HD, S] f16 (so PV B-fragments are contiguous)
__global__ __launch_bounds__(256) void vtranspose_kernel(
    const float* __restrict__ v32, _Float16* __restrict__ vt) {
  int idx = blockIdx.x * 256 + threadIdx.x;  // S*NKV*HD
  int d = idx & (HDIM - 1);
  int h = (idx >> 7) & (NKVH - 1);
  int s = idx >> 10;
  vt[((size_t)h * HDIM + d) * S_LEN + s] = (_Float16)v32[idx];
}

// ---------------- Flash attention (causal, GQA 4:1), WMMA QK^T and PV ----------------
__global__ __launch_bounds__(128) void flash_attn_kernel(
    const _Float16* __restrict__ Qh,  // [S, NH, 128]
    const _Float16* __restrict__ Kh,  // [S, NKV, 128]
    const _Float16* __restrict__ Vt,  // [NKV, 128, S]  (transposed)
    _Float16* __restrict__ Oh) {      // [S, NH*128]
  int head = blockIdx.y;
  int kvhead = head >> 2;  // NH/NKV = 4
  int wave = threadIdx.x >> 5, lane = threadIdx.x & 31;
  int half = lane >> 4, l16 = lane & 15;
  int q0 = blockIdx.x * 64 + wave * 16;  // this wave's 16 query rows

  // per-wave P staging; 16 rows x 32 halfs, columns (n, n+16) interleaved
  __shared__ alignas(16) _Float16 probs[4][16 * 32];

  // Preload Q A-fragments: HD=128 -> 4 chunks of K=32 (two 16B loads each)
  v16h qf[4];
  {
    const _Float16* qrow = Qh + ((size_t)(q0 + l16) * NHEAD + head) * HDIM;
#pragma unroll
    for (int c = 0; c < 4; ++c)
      qf[c] = cat8(*(const v8h*)(qrow + c * 32 + half * 8),
                   *(const v8h*)(qrow + c * 32 + 16 + half * 8));
  }

  v8f o[8];
#pragma unroll
  for (int df = 0; df < 8; ++df) o[df] = (v8f){};
  float mrow[8], lrow[8];
#pragma unroll
  for (int r = 0; r < 8; ++r) { mrow[r] = -1e30f; lrow[r] = 0.f; }

  const float scl = 0.08838834764831845f;  // 1/sqrt(128)
  int kv_end = q0 + 16;                    // causal: kv <= q

  for (int kv0 = 0; kv0 < kv_end; kv0 += 32) {
    // ---- scores: two 16x16 fragments over 32 kv positions ----
    v8f s0 = {}, s1 = {};
#pragma unroll
    for (int c = 0; c < 4; ++c) {
      const _Float16* kr0 = Kh + ((size_t)(kv0 + l16) * NKVH + kvhead) * HDIM + c * 32;
      const _Float16* kr1 = Kh + ((size_t)(kv0 + 16 + l16) * NKVH + kvhead) * HDIM + c * 32;
      v16h b0 = cat8(*(const v8h*)(kr0 + half * 8),
                     *(const v8h*)(kr0 + 16 + half * 8));
      v16h b1 = cat8(*(const v8h*)(kr1 + half * 8),
                     *(const v8h*)(kr1 + 16 + half * 8));
      s0 = wmma_f16(qf[c], b0, s0);
      s1 = wmma_f16(qf[c], b1, s1);
    }

    // ---- online softmax per row (row = r + 8*half, cols = lane%16 (+16)) ----
#pragma unroll
    for (int r = 0; r < 8; ++r) {
      int qrow = q0 + r + 8 * half;
      float v0 = s0[r] * scl; if (kv0 + l16 > qrow)      v0 = -1e30f;
      float v1 = s1[r] * scl; if (kv0 + 16 + l16 > qrow) v1 = -1e30f;
      float mx = fmaxf(v0, v1);
#pragma unroll
      for (int off = 8; off >= 1; off >>= 1) mx = fmaxf(mx, __shfl_xor(mx, off, 32));
      float mnew = fmaxf(mrow[r], mx);
      float alpha = __expf(mrow[r] - mnew);
      float p0 = __expf(v0 - mnew);
      float p1 = __expf(v1 - mnew);
      float rs = p0 + p1;
#pragma unroll
      for (int off = 8; off >= 1; off >>= 1) rs += __shfl_xor(rs, off, 32);
      lrow[r] = lrow[r] * alpha + rs;
      mrow[r] = mnew;
#pragma unroll
      for (int df = 0; df < 8; ++df) o[df][r] *= alpha;
      // one packed 32-bit LDS store: columns (l16, l16+16) interleaved
      v2h pp;
      pp[0] = (_Float16)p0;
      pp[1] = (_Float16)p1;
      *(v2h*)&probs[wave][(r + 8 * half) * 32 + 2 * l16] = pp;
    }
    asm volatile("s_wait_dscnt 0" ::: "memory");  // LDS RAW within wave

    // ---- reload P as A-fragment: two 16B LDS loads + register deinterleave ----
    const _Float16* pr = &probs[wave][l16 * 32 + 16 * half];
    v16h pf = deinterleave(*(const v8h*)pr, *(const v8h*)(pr + 8));

    // ---- PV: 8 d-fragments; V^T gives contiguous 16B loads per lane ----
#pragma unroll
    for (int df = 0; df < 8; ++df) {
      const _Float16* vcol =
          Vt + ((size_t)kvhead * HDIM + df * 16 + l16) * S_LEN + kv0;
      v16h vf = cat8(*(const v8h*)(vcol + half * 8),
                     *(const v8h*)(vcol + 16 + half * 8));
      o[df] = wmma_f16(pf, vf, o[df]);
    }
  }

  // ---- epilogue: normalize and store f16 attn output [S, NH*HD] ----
#pragma unroll
  for (int r = 0; r < 8; ++r) {
    int qrow = q0 + r + 8 * half;
    float inv = 1.f / lrow[r];
    _Float16* orow = Oh + (size_t)qrow * (NHEAD * HDIM) + head * HDIM + l16;
#pragma unroll
    for (int df = 0; df < 8; ++df)
      orow[df * 16] = (_Float16)(o[df][r] * inv);
  }
}

// ---------------- host launch ----------------
extern "C" void kernel_launch(void* const* d_in, const int* in_sizes, int n_in,
                              void* d_out, int out_size, void* d_ws, size_t ws_size,
                              hipStream_t stream) {
  const float* hid    = (const float*)d_in[0];
  const float* cosb   = (const float*)d_in[1];
  const float* sinb   = (const float*)d_in[2];
  const float* norm_w = (const float*)d_in[3];
  const float* wq     = (const float*)d_in[4];
  const float* wk     = (const float*)d_in[5];
  const float* wv     = (const float*)d_in[6];
  const float* wo     = (const float*)d_in[7];
  float* out = (float*)d_out;

  char* ws = (char*)d_ws;
  size_t off = 0;
  auto carve = [&](size_t bytes) -> char* {
    char* p = ws + off;
    off += (bytes + 255) & ~(size_t)255;
    return p;
  };
  _Float16* h16   = (_Float16*)carve((size_t)S_LEN * H_DIM * 2);
  float*    q32   = (float*)carve((size_t)S_LEN * NHEAD * HDIM * 4);
  float*    k32   = (float*)carve((size_t)S_LEN * NKVH * HDIM * 4);
  float*    v32   = (float*)carve((size_t)S_LEN * NKVH * HDIM * 4);
  _Float16* q16   = (_Float16*)carve((size_t)S_LEN * NHEAD * HDIM * 2);
  _Float16* k16   = (_Float16*)carve((size_t)S_LEN * NKVH * HDIM * 2);
  _Float16* vt16  = (_Float16*)carve((size_t)NKVH * HDIM * S_LEN * 2);
  _Float16* att16 = (_Float16*)carve((size_t)S_LEN * NHEAD * HDIM * 2);

  // 1) RMSNorm -> f16
  rmsnorm_kernel<<<S_LEN, 256, 0, stream>>>(hid, norm_w, h16);

  // 2) Q/K/V projections (WMMA GEMM, f32 weights converted in LDS stage)
  gemm_wmma<<<dim3((NHEAD * HDIM) / TN, S_LEN / TM), 256, 0, stream>>>(
      h16, wq, q32, nullptr, S_LEN, NHEAD * HDIM, H_DIM);
  gemm_wmma<<<dim3((NKVH * HDIM) / TN, S_LEN / TM), 256, 0, stream>>>(
      h16, wk, k32, nullptr, S_LEN, NKVH * HDIM, H_DIM);
  gemm_wmma<<<dim3((NKVH * HDIM) / TN, S_LEN / TM), 256, 0, stream>>>(
      h16, wv, v32, nullptr, S_LEN, NKVH * HDIM, H_DIM);

  // 3) RoPE on Q and K (-> f16), V -> V^T f16
  rope_kernel<<<(S_LEN * NHEAD * 64) / 256, 256, 0, stream>>>(q32, cosb, sinb, q16, NHEAD);
  rope_kernel<<<(S_LEN * NKVH * 64) / 256, 256, 0, stream>>>(k32, cosb, sinb, k16, NKVH);
  vtranspose_kernel<<<(S_LEN * NKVH * HDIM) / 256, 256, 0, stream>>>(v32, vt16);

  // 4) Flash attention (causal, GQA)
  flash_attn_kernel<<<dim3(S_LEN / 64, NHEAD), 128, 0, stream>>>(q16, k16, vt16, att16);

  // 5) Output projection + residual
  gemm_wmma<<<dim3(H_DIM / TN, S_LEN / TM), 256, 0, stream>>>(
      att16, wo, out, hid, S_LEN, H_DIM, NHEAD * HDIM);
}
